// LSTM_14628658610561
// MI455X (gfx1250) — matrix operand order repaired and checked by reading, not compiled, not run on hardware
//
#include <hip/hip_runtime.h>
#include <hip/hip_bf16.h>
#include <math.h>

#define BB 256
#define DD 128
#define TT 512
#define HH 256
#define CC 10
#define KTOT (DD + HH)     // 384
#define FOURH (4 * HH)     // 1024
#define NBLK 32            // 4 batch-tiles x 8 hidden-tiles, all co-resident
#define WPAD 392           // padded LDS row stride (elements): 784B = 49*16B, !=0 mod 64 banks

typedef __attribute__((ext_vector_type(16))) __bf16 v16bf;
typedef __attribute__((ext_vector_type(8)))  float  v8f;

__device__ __forceinline__ unsigned short f2bf(float f) {
    union { float f; unsigned u; } v; v.f = f;
    unsigned r = v.u + 0x7FFFu + ((v.u >> 16) & 1u);   // round-to-nearest-even
    return (unsigned short)(r >> 16);
}
__device__ __forceinline__ float bf2f(unsigned short s) {
    union { unsigned u; float f; } v; v.u = ((unsigned)s) << 16;
    return v.f;
}
// v_exp_f32 + v_rcp_f32 based activations (no libm branches)
__device__ __forceinline__ float fast_sigmoid(float x) {
    return __builtin_amdgcn_rcpf(1.0f + __expf(-x));
}
__device__ __forceinline__ float fast_tanh(float x) {
    return 2.0f * __builtin_amdgcn_rcpf(1.0f + __expf(-2.0f * x)) - 1.0f;
}

// async memory->LDS copy of 16B per lane (GLOBAL_LOAD_ASYNC_TO_LDS_B128, ASYNCcnt)
__device__ __forceinline__ void async_copy_b128(void* lds_dst, const void* gsrc) {
    unsigned lds_off = (unsigned)(unsigned long long)lds_dst;  // low 32b = LDS offset
    asm volatile("global_load_async_to_lds_b128 %0, %1, off"
                 :: "v"(lds_off), "v"((unsigned long long)gsrc) : "memory");
}
__device__ __forceinline__ void async_wait0() {
    asm volatile("s_wait_asynccnt 0x0" ::: "memory");
}

// ---- pack 8 weight matrices, TRANSPOSED: Wct[(g*H + j)*KTOT + k] (bf16, K contiguous)
__global__ void pack_weights_t(const float* __restrict__ Wgx, const float* __restrict__ Wix,
                               const float* __restrict__ Wfx, const float* __restrict__ Wox,
                               const float* __restrict__ Wgh, const float* __restrict__ Wih,
                               const float* __restrict__ Wfh, const float* __restrict__ Woh,
                               unsigned short* __restrict__ Wct) {
    int o = blockIdx.x * blockDim.x + threadIdx.x;
    if (o >= FOURH * KTOT) return;
    int gj = o / KTOT;
    int k  = o - gj * KTOT;
    int g  = gj >> 8;
    int j  = gj & 255;
    const float* Wx[4] = {Wgx, Wix, Wfx, Wox};
    const float* Wh[4] = {Wgh, Wih, Wfh, Woh};
    float v = (k < DD) ? Wx[g][k * HH + j] : Wh[g][(k - DD) * HH + j];
    Wct[o] = f2bf(v);
}

// ---- x [B,D,T] fp32 -> xs [T,B,D] bf16
__global__ void pack_x(const float* __restrict__ x, unsigned short* __restrict__ xs) {
    long e = (long)blockIdx.x * blockDim.x + threadIdx.x;
    if (e >= (long)TT * BB * DD) return;
    int d = (int)(e % DD);
    long r = e / DD;
    int b = (int)(r % BB);
    int t = (int)(r / BB);
    xs[e] = f2bf(x[((long)b * DD + d) * TT + t]);
}

__global__ void init_state(unsigned short* h0, unsigned short* h1, unsigned* cnt) {
    int e = blockIdx.x * blockDim.x + threadIdx.x;
    if (e < BB * HH) { h0[e] = 0; h1[e] = 0; }
    if (e == 0) *cnt = 0u;
}

// ---- persistent LSTM: each block owns a (64-batch x 32-hidden x 4-gate) slice.
// Weights live in LDS for all 512 steps; c-state lives in LDS; h is exchanged
// through L2 with a device-wide atomic-counter barrier between steps.
__global__ __launch_bounds__(256) void lstm_persistent(
    const unsigned short* __restrict__ xs,    // [T][B][D] bf16
    const unsigned short* __restrict__ Wct,   // [4H][KTOT] bf16 (K contiguous)
    unsigned short* hb0, unsigned short* hb1, // [B][H] bf16 ping-pong
    const float* __restrict__ bg, const float* __restrict__ bi,
    const float* __restrict__ bff, const float* __restrict__ bo,
    unsigned* cnt)
{
    __shared__ __align__(16) unsigned short sW[128 * WPAD]; // 100,352 B weight slice
    __shared__ __align__(16) unsigned short sA[64 * WPAD];  //  50,176 B A tile / gate alias
    __shared__ __align__(16) float          sC[64 * 32];    //   8,192 B resident c-state
    float* sG = (float*)sA;                                 // epilogue alias [4][64][32]

    const int tid  = threadIdx.x;
    const int lane = tid & 31;
    const int wave = tid >> 5;
    const int gt   = wave & 3;                 // wave's gate
    const int mrow = (wave >> 2) * 32;         // batch rows 0..31 or 32..63
    const int b0   = blockIdx.x * 64;
    const int n0   = blockIdx.y * 32;

    // ---- one-time: weight slice -> LDS (128 rows x 384 elems, b128 traffic)
#pragma unroll
    for (int i = 0; i < 24; ++i) {
        int idx = tid + i * 256;               // 0..6143
        int row = idx / 48;                    // 0..127  (g*32 + nn)
        int c16 = idx - row * 48;              // 0..47
        int g   = row >> 5;
        int nn  = row & 31;
        const uint4* src = (const uint4*)(Wct + (long)(g * HH + n0 + nn) * KTOT);
        ((uint4*)(sW + row * WPAD))[c16] = src[c16];
    }
    for (int e = tid; e < 64 * 32; e += 256) sC[e] = 0.0f;

    // epilogue slice (time-invariant): row erow, cols ecol..ecol+7
    const int erow = tid >> 2;
    const int ecol = (tid & 3) * 8;
    float rbias[4][8];
    {
        const float* bp4[4] = {bg, bi, bff, bo};
#pragma unroll
        for (int g = 0; g < 4; ++g)
#pragma unroll
            for (int c = 0; c < 8; ++c)
                rbias[g][c] = bp4[g][n0 + ecol + c];
    }
    __syncthreads();

    unsigned target = 0;
    for (int t = 0; t < TT; ++t) {
        const unsigned short* hin  = (t & 1) ? hb1 : hb0;
        unsigned short*       hout = (t & 1) ? hb0 : hb1;

        // ---- stage A [64][384] (stride WPAD) via async memory->LDS b128 copies:
        //      x_t cols 0..127, h cols 128..383
#pragma unroll
        for (int i = 0; i < 12; ++i) {
            int idx = tid + i * 256;           // 0..3071
            int row = idx / 48;
            int c16 = idx - row * 48;
            int col = c16 * 8;
            const unsigned short* src = (col < DD)
                ? xs + ((long)t * BB + (b0 + row)) * DD + col
                : hin + (long)(b0 + row) * HH + (col - DD);
            async_copy_b128(&sA[row * WPAD + col], src);
        }
        if (t + 1 < TT)  // global_prefetch_b8 of next timestep's x tile
            __builtin_prefetch(xs + ((long)(t + 1) * BB + (b0 + (tid & 63))) * DD, 0, 0);
        async_wait0();
        __syncthreads();

        // ---- GEMM: [64 x 384] @ [384 x (4x32)] with bf16 WMMA, no inner barriers
        v8f acc[2][2];
#pragma unroll
        for (int mt = 0; mt < 2; ++mt)
#pragma unroll
            for (int nt = 0; nt < 2; ++nt)
                acc[mt][nt] = (v8f){0.f,0.f,0.f,0.f,0.f,0.f,0.f,0.f};

        for (int k0 = 0; k0 < KTOT; k0 += 32) {
            union FA { v16bf v; uint4 q[2]; };
            FA a[2];
            const int kofs = (lane < 16) ? 0 : 8;   // A: K {0..7,16..23} | {8..15,24..31}
#pragma unroll
            for (int mt = 0; mt < 2; ++mt) {
                const unsigned short* pa =
                    sA + (mrow + mt * 16 + (lane & 15)) * WPAD + k0 + kofs;
                a[mt].q[0] = *(const uint4*)pa;
                a[mt].q[1] = *(const uint4*)(pa + 16);
            }
            const int kb = (lane < 16) ? 0 : 8;     // B: K {0..7,16..23} | {8..15,24..31}
#pragma unroll
            for (int nt = 0; nt < 2; ++nt) {
                FA b;
                const unsigned short* pb =
                    sW + (gt * 32 + nt * 16 + (lane & 15)) * WPAD + k0 + kb;
                b.q[0] = *(const uint4*)pb;
                b.q[1] = *(const uint4*)(pb + 16);
#pragma unroll
                for (int mt = 0; mt < 2; ++mt)
                    acc[mt][nt] = __builtin_amdgcn_wmma_f32_16x16x32_bf16(
                        false, a[mt].v, false, b.v, (short)0, acc[mt][nt], false, false);
            }
        }
        __syncthreads();   // all fragment reads of sA done before aliasing as gates

        // ---- dump gate pre-activations: row = r + (lane>=16 ? 8:0), col = lane&15
#pragma unroll
        for (int mt = 0; mt < 2; ++mt)
#pragma unroll
            for (int nt = 0; nt < 2; ++nt)
#pragma unroll
                for (int r = 0; r < 8; ++r) {
                    int row = mrow + mt * 16 + r + ((lane >= 16) ? 8 : 0);
                    int col = nt * 16 + (lane & 15);
                    sG[(gt * 64 + row) * 32 + col] = acc[mt][nt][r];
                }
        __syncthreads();

        // ---- cell update: 8 consecutive cols per thread, packed 16B h store
        union HPK { unsigned short s[8]; uint4 q; } hp;
#pragma unroll
        for (int c = 0; c < 8; ++c) {
            int col = ecol + c;
            float gp = sG[(0 * 64 + erow) * 32 + col] + rbias[0][c];
            float ip = sG[(1 * 64 + erow) * 32 + col] + rbias[1][c];
            float fp = sG[(2 * 64 + erow) * 32 + col] + rbias[2][c];
            float op = sG[(3 * 64 + erow) * 32 + col] + rbias[3][c];
            float g  = fast_tanh(gp);
            float ii = fast_sigmoid(ip);
            float ff = fast_sigmoid(fp);
            float oo = fast_sigmoid(op);
            float cn = g * ii + sC[erow * 32 + col] * ff;
            sC[erow * 32 + col] = cn;
            hp.s[c] = f2bf(fast_tanh(cn) * oo);
        }
        *(uint4*)(hout + (long)(b0 + erow) * HH + n0 + ecol) = hp.q;

        // ---- device-wide step barrier (32 co-resident blocks)
        target += NBLK;
        __threadfence();
        __syncthreads();
        if (tid == 0) {
            atomicAdd(cnt, 1u);
            while (atomicAdd(cnt, 0u) < target) __builtin_amdgcn_s_sleep(2);
        }
        __syncthreads();
        __threadfence();
    }
}

// ---- final projection out[b][j] = h[b,:] @ W_ph[:,j] + b_p[j]  (tiny)
__global__ void proj(const unsigned short* __restrict__ h,
                     const float* __restrict__ Wp, const float* __restrict__ bp,
                     float* __restrict__ out) {
    int b = blockIdx.x;
    int j = threadIdx.x;
    if (j >= CC) return;
    float s = bp[j];
    for (int k = 0; k < HH; ++k)
        s += bf2f(h[b * HH + k]) * Wp[k * CC + j];
    out[b * CC + j] = s;
}

extern "C" void kernel_launch(void* const* d_in, const int* in_sizes, int n_in,
                              void* d_out, int out_size, void* d_ws, size_t ws_size,
                              hipStream_t stream) {
    const float* x   = (const float*)d_in[0];
    const float* Wgx = (const float*)d_in[1];
    const float* Wix = (const float*)d_in[2];
    const float* Wfx = (const float*)d_in[3];
    const float* Wox = (const float*)d_in[4];
    const float* Wgh = (const float*)d_in[5];
    const float* Wih = (const float*)d_in[6];
    const float* Wfh = (const float*)d_in[7];
    const float* Woh = (const float*)d_in[8];
    const float* bg  = (const float*)d_in[9];
    const float* bi  = (const float*)d_in[10];
    const float* bff = (const float*)d_in[11];
    const float* bo  = (const float*)d_in[12];
    const float* Wp  = (const float*)d_in[13];
    const float* bp  = (const float*)d_in[14];
    float* out = (float*)d_out;

    unsigned char* ws = (unsigned char*)d_ws;
    size_t off = 0;
    auto carve = [&](size_t bytes) -> void* {
        void* p = ws + off;
        off = (off + bytes + 255) & ~(size_t)255;
        return p;
    };
    unsigned short* Wct = (unsigned short*)carve((size_t)FOURH * KTOT * 2);   // 768 KB
    unsigned short* xs  = (unsigned short*)carve((size_t)TT * BB * DD * 2);   // 32 MB
    unsigned short* hb0 = (unsigned short*)carve((size_t)BB * HH * 2);
    unsigned short* hb1 = (unsigned short*)carve((size_t)BB * HH * 2);
    unsigned*       cnt = (unsigned*)carve(256);

    pack_weights_t<<<(FOURH * KTOT + 255) / 256, 256, 0, stream>>>(
        Wgx, Wix, Wfx, Wox, Wgh, Wih, Wfh, Woh, Wct);
    pack_x<<<(int)(((long)TT * BB * DD + 255) / 256), 256, 0, stream>>>(x, xs);
    init_state<<<(BB * HH + 255) / 256, 256, 0, stream>>>(hb0, hb1, cnt);

    lstm_persistent<<<dim3(BB / 64, HH / 32), 256, 0, stream>>>(
        xs, Wct, hb0, hb1, bg, bi, bff, bo, cnt);

    // T=512 even -> final h lives in hb0
    proj<<<BB, 32, 0, stream>>>(hb0, Wp, bp, out);
}
